// InvariantBlock_91233695302362
// MI455X (gfx1250) — compile-verified, matrix operand order checked by reading, not run on hardware
//
#include <hip/hip_runtime.h>
#include <math.h>

// ---------------------------------------------------------------------------
// FourCastNet-style block (AFNO + Galerkin attention + FFN) for MI455X/gfx1250.
// All heavy math runs on v_wmma_f32_16x16x32_f16 through one universal
// complex-capable tiled GEMM, compile-time specialized so the WMMA inner loop
// is branch-free. Interior tile staging uses CDNA5 async global->LDS copies
// (global_load_async_to_lds_b128 + s_wait_asynccnt). DFTs are GEMMs against
// twiddle tables. Workspace: ~300 MB of d_ws (phase A/B regions overlap).
// ---------------------------------------------------------------------------

#define USE_ASYNC_LDS 1

typedef _Float16 half_t;
typedef __attribute__((ext_vector_type(16))) _Float16 v16h;
typedef __attribute__((ext_vector_type(8)))  _Float16 v8h;
typedef __attribute__((ext_vector_type(8)))  float    v8f;

static constexpr int TM  = 128;     // block tile M
static constexpr int TN  = 64;      // block tile N
static constexpr int TKK = 32;      // K step == WMMA K
static constexpr int LDP = TKK + 8; // padded LDS row (80B, keeps 16B align)
static constexpr int NTHREADS = 256; // 8 waves (wave32)

__device__ inline v8f wmma16(v16h a, v16h b, v8f c) {
  // v_wmma_f32_16x16x32_f16 : D = A*B + C (probe-confirmed 8-arg signature)
  return __builtin_amdgcn_wmma_f32_16x16x32_f16(false, a, false, b,
                                                (short)0, c, false, false);
}

#if USE_ASYNC_LDS
// CDNA5 async DMA: 16B global -> LDS per lane, no VGPR round-trip.
// VDST = LDS byte address (low 32 bits of generic shared pointer),
// VADDR = 64-bit global address, GV mode (saddr = off). Tracked by ASYNCcnt.
__device__ inline void async_copy16(void* ldsDst, const void* gsrc) {
  unsigned lds = (unsigned)(unsigned long long)ldsDst;
  unsigned long long ga = (unsigned long long)gsrc;
  asm volatile("global_load_async_to_lds_b128 %0, %1, off"
               :: "v"(lds), "v"(ga)
               : "memory");
}
__device__ inline void wait_async_lds() {
  asm volatile("s_wait_asynccnt 0" ::: "memory");
}
#endif

__device__ inline float apply_act(float v, int act) {
  if (act == 1) return v > 0.f ? v : 0.f;                                  // relu
  if (act == 2) return 0.5f * v * (1.f + erff(v * 0.70710678118654752f));  // gelu
  if (act == 3) {                                                          // softshrink
    const float L = 0.01f;
    return v > L ? v - L : (v < -L ? v + L : 0.f);
  }
  return v;
}

// Fragment load from LDS tile stored as rows of K (A: [m][k], B^T: [n][k]).
// ISA 7.12.2 16-bit A layout: lanes 0-15 hold K 0..7 & 16..23, lanes 16-31
// hold K 8..15 & 24..31 -> two 16-byte LDS loads per lane.
__device__ inline v16h load_frag(const half_t* p0, int lane) {
  const half_t* p = p0 + (lane & 15) * LDP;
  const int hi = lane >> 4;
  v8h a = *reinterpret_cast<const v8h*>(p + (hi ? 8 : 0));
  v8h b = *reinterpret_cast<const v8h*>(p + (hi ? 24 : 16));
  return __builtin_shufflevector(a, b, 0,1,2,3,4,5,6,7,8,9,10,11,12,13,14,15);
}

// Stage A tile into LDS [TM][LDP]; AT=0: A is [M,K]; AT=1: A is [K,M].
// Block-uniform interior/edge split keeps the guarded code off the hot path.
template <int AT>
__device__ inline void stage_tileA(half_t dst[TM][LDP], const half_t* __restrict__ A,
                                   int lda, int m0, int k0, int M, int K, int tid) {
  constexpr int CH = (TM * TKK) / 8;  // 512 vec8 chunks
  const bool interior = (m0 + TM <= M) && (k0 + TKK <= K);
  if (AT == 0) {
    if (interior) {
      for (int q = tid; q < CH; q += NTHREADS) {
        int r = q >> 2, ko = (q & 3) << 3;
        const half_t* src = A + (long)(m0 + r) * lda + (k0 + ko);
#if USE_ASYNC_LDS
        async_copy16(&dst[r][ko], src);
#else
        *reinterpret_cast<v8h*>(&dst[r][ko]) = *reinterpret_cast<const v8h*>(src);
#endif
      }
    } else {
      for (int q = tid; q < CH; q += NTHREADS) {
        int r = q >> 2, ko = (q & 3) << 3;
        int gm = m0 + r, gk = k0 + ko;
        v8h val;
        #pragma unroll
        for (int j = 0; j < 8; ++j) val[j] = (half_t)0.f;
        if (gm < M) {
          if (gk + 8 <= K) {
            val = *reinterpret_cast<const v8h*>(A + (long)gm * lda + gk);
          } else {
            for (int j = 0; j < 8; ++j)
              if (gk + j < K) val[j] = A[(long)gm * lda + gk + j];
          }
        }
        *reinterpret_cast<v8h*>(&dst[r][ko]) = val;
      }
    }
  } else {  // A is [K,M]: 128-bit load along contiguous M, scatter into LDS
    if (interior) {
      for (int q = tid; q < CH; q += NTHREADS) {
        int kr = q >> 4, mo = (q & 15) << 3;
        v8h val = *reinterpret_cast<const v8h*>(A + (long)(k0 + kr) * lda + (m0 + mo));
        #pragma unroll
        for (int j = 0; j < 8; ++j) dst[mo + j][kr] = val[j];
      }
    } else {
      for (int q = tid; q < CH; q += NTHREADS) {
        int kr = q >> 4, mo = (q & 15) << 3;
        int gk = k0 + kr, gm0 = m0 + mo;
        if (gk < K && gm0 + 8 <= M) {
          v8h val = *reinterpret_cast<const v8h*>(A + (long)gk * lda + gm0);
          #pragma unroll
          for (int j = 0; j < 8; ++j) dst[mo + j][kr] = val[j];
        } else {
          for (int j = 0; j < 8; ++j) {
            int gm = gm0 + j;
            dst[mo + j][kr] = (gk < K && gm < M) ? A[(long)gk * lda + gm] : (half_t)0.f;
          }
        }
      }
    }
  }
}

// Stage B tile into LDS as B^T [TN][LDP]; BT=1: B is [N,K]; BT=0: B is [K,N].
template <int BT>
__device__ inline void stage_tileB(half_t dst[TN][LDP], const half_t* __restrict__ B,
                                   int ldb, int n0, int k0, int N, int K, int tid) {
  constexpr int CH = (TN * TKK) / 8;  // 256 vec8 chunks
  const bool interior = (n0 + TN <= N) && (k0 + TKK <= K);
  if (BT == 1) {
    if (interior) {
      for (int q = tid; q < CH; q += NTHREADS) {
        int r = q >> 2, ko = (q & 3) << 3;
        const half_t* src = B + (long)(n0 + r) * ldb + (k0 + ko);
#if USE_ASYNC_LDS
        async_copy16(&dst[r][ko], src);
#else
        *reinterpret_cast<v8h*>(&dst[r][ko]) = *reinterpret_cast<const v8h*>(src);
#endif
      }
    } else {
      for (int q = tid; q < CH; q += NTHREADS) {
        int r = q >> 2, ko = (q & 3) << 3;
        int gn = n0 + r, gk = k0 + ko;
        v8h val;
        #pragma unroll
        for (int j = 0; j < 8; ++j) val[j] = (half_t)0.f;
        if (gn < N) {
          if (gk + 8 <= K) {
            val = *reinterpret_cast<const v8h*>(B + (long)gn * ldb + gk);
          } else {
            for (int j = 0; j < 8; ++j)
              if (gk + j < K) val[j] = B[(long)gn * ldb + gk + j];
          }
        }
        *reinterpret_cast<v8h*>(&dst[r][ko]) = val;
      }
    }
  } else {  // B is [K,N]: 128-bit load along contiguous N, scatter into LDS
    if (interior) {
      for (int q = tid; q < CH; q += NTHREADS) {
        int kr = q >> 3, no = (q & 7) << 3;
        v8h val = *reinterpret_cast<const v8h*>(B + (long)(k0 + kr) * ldb + (n0 + no));
        #pragma unroll
        for (int j = 0; j < 8; ++j) dst[no + j][kr] = val[j];
      }
    } else {
      for (int q = tid; q < CH; q += NTHREADS) {
        int kr = q >> 3, no = (q & 7) << 3;
        int gk = k0 + kr, gn0 = n0 + no;
        if (gk < K && gn0 + 8 <= N) {
          v8h val = *reinterpret_cast<const v8h*>(B + (long)gk * ldb + gn0);
          #pragma unroll
          for (int j = 0; j < 8; ++j) dst[no + j][kr] = val[j];
        } else {
          for (int j = 0; j < 8; ++j) {
            int gn = gn0 + j;
            dst[no + j][kr] = (gk < K && gn < N) ? B[(long)gk * ldb + gn] : (half_t)0.f;
          }
        }
      }
    }
  }
}

// Universal complex-capable GEMM, compile-time specialized:
//   Or = Ar*Br - Ai*Bi (+bias_r, +addin, act), Oi = Ar*Bi + Ai*Br (+bias_i, act)
// HAI/HBI: imaginary A/B present. DOI: emit Oi. AT/BT: operand layouts.
// Two-level batch: z0=z%bdiv, z1=z/bdiv with per-operand strides.
template <bool HAI, bool HBI, bool DOI, int AT, int BT>
__global__ __launch_bounds__(NTHREADS)
void cgemm_kernel(int M, int N, int K, int bdiv,
                  const half_t* __restrict__ Ar, const half_t* __restrict__ Ai,
                  int lda, long aS0, long aS1,
                  const half_t* __restrict__ Br, const half_t* __restrict__ Bi,
                  int ldb, long bS0, long bS1,
                  const float* __restrict__ biasr, const float* __restrict__ biasi,
                  const float* __restrict__ addin,
                  float* __restrict__ Orf, half_t* __restrict__ Orh,
                  float* __restrict__ Oif, half_t* __restrict__ Oih,
                  int ldc, long cS0, long cS1,
                  float scale, int act) {
  __shared__ half_t As[HAI ? 2 : 1][TM][LDP];
  __shared__ half_t Bs[HBI ? 2 : 1][TN][LDP];

  const int tid  = threadIdx.x;
  const int lane = tid & 31;
  const int wid  = tid >> 5;
  const int m0 = blockIdx.y * TM;
  const int n0 = blockIdx.x * TN;
  const int z  = blockIdx.z;
  const int z0 = z % bdiv, z1 = z / bdiv;
  const long offA = (long)z1 * aS1 + (long)z0 * aS0;
  const long offB = (long)z1 * bS1 + (long)z0 * bS0;
  const long offC = (long)z1 * cS1 + (long)z0 * cS0;

  const half_t* ArB = Ar + offA;
  const half_t* AiB = HAI ? Ai + offA : nullptr;
  const half_t* BrB = Br + offB;
  const half_t* BiB = HBI ? Bi + offB : nullptr;

  v8f accR[2][2], accI[2][2];
  #pragma unroll
  for (int i = 0; i < 2; ++i)
    #pragma unroll
    for (int j = 0; j < 2; ++j)
      #pragma unroll
      for (int e = 0; e < 8; ++e) { accR[i][j][e] = 0.f; accI[i][j][e] = 0.f; }

  const int wm = (wid & 3) * 32;   // 4 waves along M
  const int wn = (wid >> 2) * 32;  // 2 waves along N

  for (int k0 = 0; k0 < K; k0 += TKK) {
    stage_tileA<AT>(As[0], ArB, lda, m0, k0, M, K, tid);
    if (HAI) stage_tileA<AT>(As[HAI ? 1 : 0], AiB, lda, m0, k0, M, K, tid);
    stage_tileB<BT>(Bs[0], BrB, ldb, n0, k0, N, K, tid);
    if (HBI) stage_tileB<BT>(Bs[HBI ? 1 : 0], BiB, ldb, n0, k0, N, K, tid);
    if (k0 + TKK < K) {  // hint next tiles toward the caches (global_prefetch_b8)
      __builtin_prefetch(ArB + (AT ? (long)(k0 + TKK) * lda + m0
                                   : (long)m0 * lda + k0 + TKK), 0, 0);
      __builtin_prefetch(BrB + (BT ? (long)n0 * ldb + k0 + TKK
                                   : (long)(k0 + TKK) * ldb + n0), 0, 0);
    }
#if USE_ASYNC_LDS
    wait_async_lds();   // ASYNCcnt not covered by __syncthreads (ISA 7.1)
#endif
    __syncthreads();

    // Branch-free WMMA block: up to 16 v_wmma per K-step in the complex case.
    v16h ar[2], ai[2], brf[2], bif[2], bin[2];
    #pragma unroll
    for (int i = 0; i < 2; ++i) {
      ar[i] = load_frag(&As[0][wm + 16 * i][0], lane);
      if (HAI) ai[i] = load_frag(&As[HAI ? 1 : 0][wm + 16 * i][0], lane);
    }
    #pragma unroll
    for (int j = 0; j < 2; ++j) {
      brf[j] = load_frag(&Bs[0][wn + 16 * j][0], lane);
      if (HBI) { bif[j] = load_frag(&Bs[HBI ? 1 : 0][wn + 16 * j][0], lane); bin[j] = -bif[j]; }
    }
    #pragma unroll
    for (int i = 0; i < 2; ++i)
      #pragma unroll
      for (int j = 0; j < 2; ++j) {
        accR[i][j] = wmma16(ar[i], brf[j], accR[i][j]);
        if (HAI && HBI) accR[i][j] = wmma16(ai[i], bin[j], accR[i][j]);  // -Ai*Bi
        if (DOI) {
          if (HAI) accI[i][j] = wmma16(ai[i], brf[j], accI[i][j]);
          if (HBI) accI[i][j] = wmma16(ar[i], bif[j], accI[i][j]);
        }
      }
    __syncthreads();
  }

  // Epilogue. C layout (ISA 7.12.2): VGPR e -> M=e (lanes 0-15) / e+8 (16-31),
  // N = lane&15.
  const int mh = (lane >= 16) ? 8 : 0;
  const int nl = lane & 15;
  #pragma unroll
  for (int i = 0; i < 2; ++i)
    #pragma unroll
    for (int j = 0; j < 2; ++j)
      #pragma unroll
      for (int e = 0; e < 8; ++e) {
        int m = m0 + wm + 16 * i + e + mh;
        int n = n0 + wn + 16 * j + nl;
        if (m < M && n < N) {
          long idx = offC + (long)m * ldc + n;
          float vr = accR[i][j][e] * scale;
          if (biasr) vr += biasr[n];
          if (addin) vr += addin[idx];
          vr = apply_act(vr, act);
          if (Orf) Orf[idx] = vr;
          if (Orh) Orh[idx] = (half_t)vr;
          if (DOI) {
            float vi = accI[i][j][e] * scale;
            if (biasi) vi += biasi[n];
            vi = apply_act(vi, act);
            if (Oif) Oif[idx] = vi;
            if (Oih) Oih[idx] = (half_t)vi;
          }
        }
      }
}

// LayerNorm over C=256: one wave per token, 8 channels/lane, shfl reduction.
__global__ __launch_bounds__(256)
void ln256_kernel(const float* __restrict__ xin, const float* __restrict__ g,
                  const float* __restrict__ b, float* __restrict__ outf,
                  half_t* __restrict__ outh, int addInput, int ntok) {
  const int lane = threadIdx.x & 31, wid = threadIdx.x >> 5;
  const long tok = (long)blockIdx.x * 8 + wid;
  if (tok >= ntok) return;
  const float* xp = xin + tok * 256;
  float v[8], s = 0.f;
  #pragma unroll
  for (int j = 0; j < 8; ++j) { v[j] = xp[lane + 32 * j]; s += v[j]; }
  for (int o = 16; o; o >>= 1) s += __shfl_xor(s, o, 32);
  const float mu = s * (1.f / 256.f);
  float var = 0.f;
  #pragma unroll
  for (int j = 0; j < 8; ++j) { float d = v[j] - mu; var += d * d; }
  for (int o = 16; o; o >>= 1) var += __shfl_xor(var, o, 32);
  const float r = rsqrtf(var * (1.f / 256.f) + 1e-5f);
  #pragma unroll
  for (int j = 0; j < 8; ++j) {
    int c = lane + 32 * j;
    float y = (v[j] - mu) * r * g[c] + b[c];
    if (outf) outf[tok * 256 + c] = addInput ? y + v[j] : y;
    if (outh) outh[tok * 256 + c] = (half_t)y;
  }
}

// Per-head LayerNorm over DH=32 (lane == channel), one wave per (token,head).
__global__ __launch_bounds__(256)
void headln_kernel(const half_t* __restrict__ xin, const float* __restrict__ g,
                   const float* __restrict__ b, half_t* __restrict__ out, int ntok) {
  const int lane = threadIdx.x & 31, wid = threadIdx.x >> 5;
  const long pair = (long)blockIdx.x * 8 + wid;
  if (pair >= (long)ntok * 8) return;
  const int head = (int)(pair & 7);
  const long tok = pair >> 3;
  const long idx = tok * 256 + head * 32 + lane;
  float v = (float)xin[idx];
  float mu = v;
  for (int o = 16; o; o >>= 1) mu += __shfl_xor(mu, o, 32);
  mu *= (1.f / 32.f);
  float d = v - mu, var = d * d;
  for (int o = 16; o; o >>= 1) var += __shfl_xor(var, o, 32);
  var *= (1.f / 32.f);
  const float y = d * rsqrtf(var + 1e-5f) * g[head * 32 + lane] + b[head * 32 + lane];
  out[idx] = (half_t)y;
}

__global__ void f32_to_f16_kernel(half_t* __restrict__ dst,
                                  const float* __restrict__ src, int n) {
  int i = blockIdx.x * blockDim.x + threadIdx.x;
  if (i < n) dst[i] = (half_t)src[i];
}

// Twiddle tables (ortho scale 1/sqrt(128) per direction folded in).
__global__ void build_tables_kernel(half_t* cosW, half_t* sinWneg,
                                    half_t* cosH, half_t* sinH, half_t* sinHneg,
                                    half_t* a2r, half_t* a2s) {
  const int idx = blockIdx.x * blockDim.x + threadIdx.x;
  const float s  = 0.0883883476483184406f;               // 1/sqrt(128)
  const float w0 = 6.28318530717958647692f / 128.f;
  if (idx < 80 * 128) {  // forward W-rDFT: rows kw (pad to 80), cols w
    int kw = idx / 128, w = idx % 128;
    float sv, cv; __sincosf(w0 * (float)((kw * w) & 127), &sv, &cv);
    cosW[idx]    = (half_t)(kw < 65 ?  cv * s : 0.f);
    sinWneg[idx] = (half_t)(kw < 65 ? -sv * s : 0.f);
  }
  if (idx < 128 * 128) { // H DFT tables (symmetric in fwd/inv)
    int kh = idx / 128, h = idx % 128;
    float sv, cv; __sincosf(w0 * (float)((kh * h) & 127), &sv, &cv);
    cosH[idx]    = (half_t)( cv * s);
    sinH[idx]    = (half_t)( sv * s);
    sinHneg[idx] = (half_t)(-sv * s);
  }
  if (idx < 128 * 80) {  // inverse W-rDFT: rows w, cols kw (hermitian weights)
    int w = idx / 80, kw = idx % 80;
    if (kw < 65) {
      float wgt = (kw == 0 || kw == 64) ? 1.f : 2.f;
      float sv, cv; __sincosf(w0 * (float)((kw * w) & 127), &sv, &cv);
      a2r[idx] = (half_t)(wgt * cv * s);
      a2s[idx] = (half_t)(wgt * sv * s);
    } else {
      a2r[idx] = (half_t)0.f; a2s[idx] = (half_t)0.f;
    }
  }
}

// ---------------------------------------------------------------------------
extern "C" void kernel_launch(void* const* d_in, const int* in_sizes, int n_in,
                              void* d_out, int out_size, void* d_ws, size_t ws_size,
                              hipStream_t stream) {
  (void)in_sizes; (void)n_in; (void)out_size; (void)ws_size;

  const float* x     = (const float*)d_in[0];
  const float* w1    = (const float*)d_in[1];   // [2,16,16,128]
  const float* b1    = (const float*)d_in[2];   // [2,16,128]
  const float* w2    = (const float*)d_in[3];   // [2,16,128,16]
  const float* b2    = (const float*)d_in[4];   // [2,16,16]
  const float* wq    = (const float*)d_in[5];
  const float* bq    = (const float*)d_in[6];
  const float* wk    = (const float*)d_in[7];
  const float* bk    = (const float*)d_in[8];
  const float* wv    = (const float*)d_in[9];
  const float* bv    = (const float*)d_in[10];
  const float* lnk_g = (const float*)d_in[11];
  const float* lnk_b = (const float*)d_in[12];
  const float* lnv_g = (const float*)d_in[13];
  const float* lnv_b = (const float*)d_in[14];
  const float* wo    = (const float*)d_in[15];
  const float* bo    = (const float*)d_in[16];
  const float* w_up  = (const float*)d_in[17];
  const float* b_up  = (const float*)d_in[18];
  const float* w_dn  = (const float*)d_in[19];
  const float* b_dn  = (const float*)d_in[20];
  const float* n1g   = (const float*)d_in[21];
  const float* n1b   = (const float*)d_in[22];
  const float* n2g   = (const float*)d_in[23];
  const float* n2b   = (const float*)d_in[24];
  const float* n3g   = (const float*)d_in[25];
  const float* n3b   = (const float*)d_in[26];

  const int NTOK = 32768;                // B*H*W
  const int KW   = 65;                   // W/2+1
  const int P    = 2 * 128 * KW;         // 16640 spectral rows (mult of 16)
  const long SPC = (long)128 * KW * 256; // per-batch spectral stride 2129920

  char* ws = (char*)d_ws;
  size_t off = 0;
  auto alloc = [&](size_t bytes) -> char* {
    char* p = ws + off;
    off += (bytes + 255) & ~(size_t)255;
    return p;
  };

  // ---- persistent region (weights f16, tables, x1 spine) ----
  half_t* wqh  = (half_t*)alloc(65536 * 2);
  half_t* wkh  = (half_t*)alloc(65536 * 2);
  half_t* wvh  = (half_t*)alloc(65536 * 2);
  half_t* woh  = (half_t*)alloc(65536 * 2);
  half_t* wuph = (half_t*)alloc(524288 * 2);
  half_t* wdnh = (half_t*)alloc(524288 * 2);
  half_t* w1rh = (half_t*)alloc(32768 * 2);
  half_t* w1ih = (half_t*)alloc(32768 * 2);
  half_t* w2rh = (half_t*)alloc(32768 * 2);
  half_t* w2ih = (half_t*)alloc(32768 * 2);
  half_t* cosW    = (half_t*)alloc(80 * 128 * 2);
  half_t* sinWneg = (half_t*)alloc(80 * 128 * 2);
  half_t* cosH    = (half_t*)alloc(128 * 128 * 2);
  half_t* sinH    = (half_t*)alloc(128 * 128 * 2);
  half_t* sinHneg = (half_t*)alloc(128 * 128 * 2);
  half_t* a2r     = (half_t*)alloc(128 * 80 * 2);
  half_t* a2s     = (half_t*)alloc(128 * 80 * 2);
  float*  x1      = (float*)alloc((size_t)NTOK * 256 * 4);

  const size_t phase_base = off;

  // ---- phase A scratch (AFNO) ----
  float*  xsum = (float*)alloc((size_t)NTOK * 256 * 4);   // x + LN1(x)
  half_t* xn1h = (half_t*)alloc((size_t)NTOK * 256 * 2);
  half_t* fr1  = (half_t*)alloc((size_t)P * 256 * 2);     // W-DFT out (re)
  half_t* fi1  = (half_t*)alloc((size_t)P * 256 * 2);     // W-DFT out (im)
  half_t* xr2h = (half_t*)alloc((size_t)P * 256 * 2);     // full 2D spectrum
  half_t* xi2h = (half_t*)alloc((size_t)P * 256 * 2);
  half_t* o1rh = (half_t*)alloc((size_t)P * 128 * 2);     // per-block hidden
  half_t* o1ih = (half_t*)alloc((size_t)P * 128 * 2);
  half_t* o2rh = (half_t*)alloc((size_t)P * 256 * 2);     // post-shrink spectrum
  half_t* o2ih = (half_t*)alloc((size_t)P * 256 * 2);
  half_t* yrh  = fr1;                                     // iH output aliases fr1/fi1
  half_t* yih  = fi1;

  auto conv = [&](half_t* d, const float* s, int n) {
    f32_to_f16_kernel<<<dim3((n + 255) / 256), 256, 0, stream>>>(d, s, n);
  };
  // Dispatch to the compile-time specialized kernels.
  auto cg = [&](int M, int N, int K, int batch, int bdiv,
                const half_t* Ar, const half_t* Ai, int lda, int at, long aS0, long aS1,
                const half_t* Br, const half_t* Bi, int ldb, int bt, long bS0, long bS1,
                const float* br_, const float* bi_, const float* add_,
                float* Orf, half_t* Orh, float* Oif, half_t* Oih,
                int ldc, long cS0, long cS1, float scale, int act) {
    dim3 g((N + TN - 1) / TN, (M + TM - 1) / TM, batch);
    const bool hai = (Ai != nullptr), hbi = (Bi != nullptr);
    const bool doi = (Oif != nullptr) || (Oih != nullptr);
#define LAUNCH(HA, HB, DO, ATT, BTT)                                          \
    cgemm_kernel<HA, HB, DO, ATT, BTT><<<g, NTHREADS, 0, stream>>>(           \
        M, N, K, bdiv, Ar, Ai, lda, aS0, aS1, Br, Bi, ldb, bS0, bS1,          \
        br_, bi_, add_, Orf, Orh, Oif, Oih, ldc, cS0, cS1, scale, act)
    if (!hai && !hbi) {
      if (at)      LAUNCH(false, false, false, 1, 0);   // kv = K^T V
      else if (bt) LAUNCH(false, false, false, 0, 1);   // dense x weight^T
      else         LAUNCH(false, false, false, 0, 0);   // attn = Q kv
    } else if (hai && !hbi) {
      if (doi) LAUNCH(true, false, true, 0, 0);         // W-rDFT (real input)
      else     LAUNCH(true, false, false, 0, 0);
    } else {
      if (doi) LAUNCH(true, true, true, 0, 0);          // full complex GEMM
      else     LAUNCH(true, true, false, 0, 0);         // iW-rDFT (real output)
    }
#undef LAUNCH
  };

  // ---- prep: twiddle tables + f16 weights ----
  build_tables_kernel<<<64, 256, 0, stream>>>(cosW, sinWneg, cosH, sinH, sinHneg, a2r, a2s);
  conv(wqh, wq, 65536);  conv(wkh, wk, 65536);
  conv(wvh, wv, 65536);  conv(woh, wo, 65536);
  conv(wuph, w_up, 524288); conv(wdnh, w_dn, 524288);
  conv(w1rh, w1, 32768); conv(w1ih, w1 + 32768, 32768);
  conv(w2rh, w2, 32768); conv(w2ih, w2 + 32768, 32768);

  // ---- 1) AFNO block ----
  // LN1: xsum = x + LN(x) (AFNO residual spine), xn1h = f16(LN(x))
  ln256_kernel<<<NTOK / 8, 256, 0, stream>>>(x, n1g, n1b, xsum, xn1h, 1, NTOK);

  // W-direction rDFT: per (b,h): [65x256] = Twiddle[65x128] @ X[128x256]
  cg(65, 256, 128, 256, 256,
     cosW, sinWneg, 128, 0, 0, 0,
     xn1h, nullptr, 256, 0, 32768, 0,
     nullptr, nullptr, nullptr,
     nullptr, fr1, nullptr, fi1, 256, (long)KW * 256, 0, 1.f, 0);

  // H-direction complex DFT: per (b,kw) over kh
  cg(128, 256, 128, 130, KW,
     cosH, sinHneg, 128, 0, 0, 0,
     fr1, fi1, KW * 256, 0, 256, SPC,
     nullptr, nullptr, nullptr,
     nullptr, xr2h, nullptr, xi2h, KW * 256, 256, SPC, 1.f, 0);

  // block-diagonal complex MLP: relu(X W1 + b1) -> shrink(H W2 + b2)
  for (int n = 0; n < 16; ++n) {
    cg(P, 128, 16, 1, 1,
       xr2h + n * 16, xi2h + n * 16, 256, 0, 0, 0,
       w1rh + n * 2048, w1ih + n * 2048, 128, 0, 0, 0,
       b1 + n * 128, b1 + 2048 + n * 128, nullptr,
       nullptr, o1rh, nullptr, o1ih, 128, 0, 0, 1.f, 1);
    cg(P, 16, 128, 1, 1,
       o1rh, o1ih, 128, 0, 0, 0,
       w2rh + n * 2048, w2ih + n * 2048, 16, 0, 0, 0,
       b2 + n * 16, b2 + 256 + n * 16, nullptr,
       nullptr, o2rh + n * 16, nullptr, o2ih + n * 16, 256, 0, 0, 1.f, 3);
  }

  // inverse H DFT (e^{+i}): per (b,kw)
  cg(128, 256, 128, 130, KW,
     cosH, sinH, 128, 0, 0, 0,
     o2rh, o2ih, KW * 256, 0, 256, SPC,
     nullptr, nullptr, nullptr,
     nullptr, yrh, nullptr, yih, KW * 256, 256, SPC, 1.f, 0);

  // inverse W rDFT (hermitian weights) + residual -> x1 = x + LN1(x) + irfft2
  cg(128, 256, KW, 256, 256,
     a2r, a2s, 80, 0, 0, 0,
     yrh, yih, 256, 0, (long)KW * 256, 0,
     nullptr, nullptr, xsum,
     x1, nullptr, nullptr, nullptr, 256, 32768, 0, 1.f, 0);

  // ---- phase B scratch (attention + FFN) overlaps phase A ----
  off = phase_base;
  float*  x2n   = (float*)alloc((size_t)NTOK * 256 * 4);
  half_t* x2nh  = (half_t*)alloc((size_t)NTOK * 256 * 2);
  half_t* qh    = (half_t*)alloc((size_t)NTOK * 256 * 2);
  half_t* kraw  = (half_t*)alloc((size_t)NTOK * 256 * 2);
  half_t* vraw  = (half_t*)alloc((size_t)NTOK * 256 * 2);
  half_t* khh   = (half_t*)alloc((size_t)NTOK * 256 * 2);
  half_t* vhh   = (half_t*)alloc((size_t)NTOK * 256 * 2);
  half_t* kvh   = (half_t*)alloc(16 * 1024 * 2);
  half_t* attnh = (half_t*)alloc((size_t)NTOK * 256 * 2);
  float*  x3    = (float*)alloc((size_t)NTOK * 256 * 4);
  float*  x3n   = (float*)alloc((size_t)NTOK * 256 * 4);
  half_t* x3nh  = (half_t*)alloc((size_t)NTOK * 256 * 2);
  half_t* hmidh = (half_t*)alloc((size_t)NTOK * 512 * 2);  // FFN chunk buffer

  // ---- 2) Galerkin linear attention ----
  ln256_kernel<<<NTOK / 8, 256, 0, stream>>>(x1, n2g, n2b, x2n, x2nh, 0, NTOK);

  cg(NTOK, 256, 256, 1, 1, x2nh, nullptr, 256, 0, 0, 0,
     wqh, nullptr, 256, 1, 0, 0, bq, nullptr, nullptr,
     nullptr, qh, nullptr, nullptr, 256, 0, 0, 1.f, 0);
  cg(NTOK, 256, 256, 1, 1, x2nh, nullptr, 256, 0, 0, 0,
     wkh, nullptr, 256, 1, 0, 0, bk, nullptr, nullptr,
     nullptr, kraw, nullptr, nullptr, 256, 0, 0, 1.f, 0);
  cg(NTOK, 256, 256, 1, 1, x2nh, nullptr, 256, 0, 0, 0,
     wvh, nullptr, 256, 1, 0, 0, bv, nullptr, nullptr,
     nullptr, vraw, nullptr, nullptr, 256, 0, 0, 1.f, 0);

  headln_kernel<<<NTOK, 256, 0, stream>>>(kraw, lnk_g, lnk_b, khh, NTOK);
  headln_kernel<<<NTOK, 256, 0, stream>>>(vraw, lnv_g, lnv_b, vhh, NTOK);

  // kv[b,h] = K^T V / N : M=32, N=32, K=16384, batch (b,head)
  cg(32, 32, 16384, 16, 8,
     khh, nullptr, 256, 1 /*A is [K,M]*/, 32, (long)16384 * 256,
     vhh, nullptr, 256, 0, 32, (long)16384 * 256,
     nullptr, nullptr, nullptr,
     nullptr, kvh, nullptr, nullptr, 32, 1024, 8192,
     1.f / 16384.f, 0);

  // attn = Q @ kv : per (b,head)
  cg(16384, 32, 32, 16, 8,
     qh, nullptr, 256, 0, 32, (long)16384 * 256,
     kvh, nullptr, 32, 0, 1024, 8192,
     nullptr, nullptr, nullptr,
     nullptr, attnh, nullptr, nullptr, 256, 32, (long)16384 * 256, 1.f, 0);

  // out-proj + residual (vs LN2 output) -> x3
  cg(NTOK, 256, 256, 1, 1, attnh, nullptr, 256, 0, 0, 0,
     woh, nullptr, 256, 1, 0, 0, bo, nullptr, x2n,
     x3, nullptr, nullptr, nullptr, 256, 0, 0, 1.f, 0);

  // ---- 3) FFN (1x1 conv) ----
  ln256_kernel<<<NTOK / 8, 256, 0, stream>>>(x3, n3g, n3b, x3n, x3nh, 0, NTOK);

  float* outp = (float*)d_out;
  for (int c = 0; c < 4; ++c) {  // 4 x N=512 chunks, accumulate into d_out
    cg(NTOK, 512, 256, 1, 1,
       x3nh, nullptr, 256, 0, 0, 0,
       wuph + (long)c * 512 * 256, nullptr, 256, 1, 0, 0,
       b_up + c * 512, nullptr, nullptr,
       nullptr, hmidh, nullptr, nullptr, 512, 0, 0, 1.f, 2 /*gelu*/);
    cg(NTOK, 256, 512, 1, 1,
       hmidh, nullptr, 512, 0, 0, 0,
       wdnh + c * 512, nullptr, 2048, 1, 0, 0,
       (c == 3) ? b_dn : nullptr, nullptr, (c == 0) ? x3n : outp,
       outp, nullptr, nullptr, nullptr, 256, 0, 0, 1.f, 0);
  }
}
